// CustomMTCNN_8194797600834
// MI455X (gfx1250) — compile-verified, matrix operand order checked by reading, not run on hardware
//
#include <hip/hip_runtime.h>

typedef __attribute__((ext_vector_type(16))) _Float16 v16h;
typedef __attribute__((ext_vector_type(8)))  float    v8f;

#define K_TOP   1024
#define THR0_C  0.6f
#define NMS_THR 0.7f

static __device__ __forceinline__ v8f wmma16x16x32(v16h a, v16h b, v8f c) {
  // 8 args: (neg_a, A, neg_b, B, c_mod, C, reuse_a, reuse_b)
  return __builtin_amdgcn_wmma_f32_16x16x32_f16(false, a, false, b, (short)0, c, false, false);
}

// ---------------------------------------------------------------------------
// Weight pre-pack: lay conv weights out in the exact per-lane B-fragment order
// for V_WMMA_F32_16X16X32_F16 so each lane loads one contiguous v16h (32B).
// K order = (ky*3+kx)*Cin + ci.  Layout: [ntile][kchunk][lane 0..31][half 0..15]
// ---------------------------------------------------------------------------
__global__ void pack_conv_b(const float* __restrict__ w, _Float16* __restrict__ out,
                            int Cin, int Cout, int ntiles, int kchunks) {
  int e = blockIdx.x * blockDim.x + threadIdx.x;
  int total = ntiles * kchunks * 512;
  if (e >= total) return;
  int h = e & 15, l = (e >> 4) & 31, rest = e >> 9;
  int kc = rest % kchunks, nt = rest / kchunks;
  int N = nt * 16 + (l & 15);
  int v = h >> 1, pos = h & 1;
  int Kb = (l & 16) ? 8 : 0;
  int Kl = Kb + ((v < 4) ? 2 * v : 16 + 2 * (v - 4)) + pos;
  int K = kc * 32 + Kl;
  int Kreal = Cin * 9;
  float val = 0.f;
  if (K < Kreal && N < Cout) {
    int ci = K % Cin, tap = K / Cin;
    int ky = tap / 3, kx = tap % 3;
    val = w[((N * Cin + ci) * 3 + ky) * 3 + kx];
  }
  out[e] = (_Float16)val;
}

// 1x1 head weights: N = [cls0, cls1, reg0..reg3, 0...], K = 32 input channels.
__global__ void pack_head_b(const float* __restrict__ cls_w,
                            const float* __restrict__ reg_w,
                            _Float16* __restrict__ out) {
  int e = blockIdx.x * blockDim.x + threadIdx.x;
  if (e >= 512) return;
  int h = e & 15, l = (e >> 4) & 31;
  int N = l & 15;
  int v = h >> 1, pos = h & 1;
  int Kb = (l & 16) ? 8 : 0;
  int K = Kb + ((v < 4) ? 2 * v : 16 + 2 * (v - 4)) + pos;
  float val = 0.f;
  if (N < 2)      val = cls_w[N * 32 + K];
  else if (N < 6) val = reg_w[(N - 2) * 32 + K];
  out[e] = (_Float16)val;
}

// ---------------------------------------------------------------------------
// conv1 (3->10, 3x3) + PReLU + 2x2/2 maxpool, fused. Output NHWC f16 (C=10).
// ---------------------------------------------------------------------------
__global__ void conv1_prelu_pool(const float* __restrict__ img,
                                 const float* __restrict__ w,
                                 const float* __restrict__ bias,
                                 const float* __restrict__ pa,
                                 _Float16* __restrict__ out) {
  const int NP = 255 * 255;
  int t = blockIdx.x * blockDim.x + threadIdx.x;
  if (t >= 16 * NP) return;
  int px = t % 255, py = (t / 255) % 255, n = t / NP;
  int y0 = 2 * py, x0 = 2 * px;
  float patch[3][4][4];
#pragma unroll
  for (int ci = 0; ci < 3; ++ci)
#pragma unroll
    for (int dy = 0; dy < 4; ++dy)
#pragma unroll
      for (int dx = 0; dx < 4; ++dx)
        patch[ci][dy][dx] = img[(((size_t)n * 3 + ci) * 512 + (y0 + dy)) * 512 + (x0 + dx)];
  _Float16* op = out + (size_t)t * 10;
  for (int oc = 0; oc < 10; ++oc) {
    float b = bias[oc], a = pa[oc];
    float m = -__builtin_inff();
#pragma unroll
    for (int sy = 0; sy < 2; ++sy)
#pragma unroll
      for (int sx = 0; sx < 2; ++sx) {
        float s = b;
#pragma unroll
        for (int ci = 0; ci < 3; ++ci)
#pragma unroll
          for (int ky = 0; ky < 3; ++ky)
#pragma unroll
            for (int kx = 0; kx < 3; ++kx)
              s += patch[ci][sy + ky][sx + kx] * w[((oc * 3 + ci) * 3 + ky) * 3 + kx];
        s = s > 0.f ? s : a * s;
        m = fmaxf(m, s);
      }
    op[oc] = (_Float16)m;
  }
}

// ---------------------------------------------------------------------------
// 3x3 conv as implicit GEMM with WMMA f16->f32.
// Workgroup = 256 threads = 8 wave32; tile = 128 output pixels.
// LDS holds the im2col A-tile [128 x Kpad] halves; each wave owns 16 rows and
// computes a 16x16 (NTILES=1) or 16x32 (NTILES=2) output tile.
// Staging: 2 threads per pixel, dword-wide loads, incremental (tap,ci) walk.
// ---------------------------------------------------------------------------
template <int NTILES>
__global__ void conv3x3_wmma(const _Float16* __restrict__ in,
                             const _Float16* __restrict__ Bp,
                             const float* __restrict__ bias,
                             const float* __restrict__ pa,
                             _Float16* __restrict__ out,
                             int Hin, int Win, int Cin,
                             int Hout, int Wout, int Cout,
                             int kchunks, int Kpad) {
  extern __shared__ _Float16 Alds[];
  const int npix = Hout * Wout;
  const int tilesPerBatch = (npix + 127) >> 7;
  const int b = blockIdx.x / tilesPerBatch;
  const int pixbase = (blockIdx.x % tilesPerBatch) << 7;
  const int Kreal = Cin * 9;
  const _Float16* inb = in + (size_t)b * Hin * Win * Cin;

  {
    const int Khalf = Kpad >> 1;
    int p = threadIdx.x >> 1;
    int half = threadIdx.x & 1;
    int pix = pixbase + p; if (pix >= npix) pix = npix - 1;
    int py = pix / Wout, px = pix - py * Wout;
    const _Float16* base = inb + ((size_t)py * Win + px) * Cin;
    int k0 = half * Khalf;
    int tap = k0 / Cin;
    int ci  = k0 - tap * Cin;          // even (k0, Cin even)
    int ky  = tap / 3, kx = tap - ky * 3;
    int tapoff = (ky * Win + kx) * Cin;
    int off = tapoff + ci;
    _Float16* dst = Alds + p * Kpad + k0;
    for (int kk = 0; kk < Khalf; kk += 2) {
      unsigned val = 0;
      if (k0 + kk < Kreal) val = *(const unsigned*)(base + off);
      *(unsigned*)(dst + kk) = val;
      ci += 2; off += 2;
      if (ci >= Cin) {
        ci = 0;
        if (kx < 2) { ++kx; tapoff += Cin; }
        else        { kx = 0; ++ky; tapoff += (Win - 2) * Cin; }
        off = tapoff;
      }
    }
  }
  __syncthreads();

  const int lane = threadIdx.x & 31;
  const int wave = threadIdx.x >> 5;
  const int n0 = lane & 15;

  union U8 { v8f v; float f[8]; } acc0, acc1;
  {
    float bb0 = bias[n0];
#pragma unroll
    for (int i = 0; i < 8; ++i) acc0.f[i] = bb0;
    if (NTILES == 2) {
      float bb1 = bias[16 + n0];
#pragma unroll
      for (int i = 0; i < 8; ++i) acc1.f[i] = bb1;
    }
  }

  const unsigned* rowp =
      (const unsigned*)(Alds + (size_t)((wave << 4) + n0) * Kpad);

  for (int kc = 0; kc < kchunks; ++kc) {
    int kb = (kc << 5) + ((lane & 16) ? 8 : 0);
    union { v16h v; unsigned u[8]; } a;
    a.u[0] = rowp[(kb +  0) >> 1];
    a.u[1] = rowp[(kb +  2) >> 1];
    a.u[2] = rowp[(kb +  4) >> 1];
    a.u[3] = rowp[(kb +  6) >> 1];
    a.u[4] = rowp[(kb + 16) >> 1];
    a.u[5] = rowp[(kb + 18) >> 1];
    a.u[6] = rowp[(kb + 20) >> 1];
    a.u[7] = rowp[(kb + 22) >> 1];
    v16h b0 = *(const v16h*)(Bp + ((size_t)(kc * 32 + lane) << 4));
    acc0.v = wmma16x16x32(a.v, b0, acc0.v);
    if (NTILES == 2) {
      v16h b1 = *(const v16h*)(Bp + ((size_t)((kchunks + kc) * 32 + lane) << 4));
      acc1.v = wmma16x16x32(a.v, b1, acc1.v);
    }
  }

  float a0 = pa[n0];
  float a1 = (NTILES == 2) ? pa[16 + n0] : 0.f;
#pragma unroll
  for (int i = 0; i < 8; ++i) {
    int row = (wave << 4) + ((lane & 16) ? 8 + i : i);
    int pix = pixbase + row;
    if (pix < npix) {
      _Float16* op = out + ((size_t)b * npix + pix) * Cout;
      float d0 = acc0.f[i]; d0 = d0 > 0.f ? d0 : a0 * d0;
      op[n0] = (_Float16)d0;
      if (NTILES == 2) {
        float d1 = acc1.f[i]; d1 = d1 > 0.f ? d1 : a1 * d1;
        op[16 + n0] = (_Float16)d1;
      }
    }
  }
}

// ---------------------------------------------------------------------------
// 1x1 heads (cls 2ch + reg 4ch packed into one N=16 WMMA tile) + softmax.
// ---------------------------------------------------------------------------
__global__ void head_softmax(const _Float16* __restrict__ x3,
                             const _Float16* __restrict__ Bh,
                             const float* __restrict__ cls_b,
                             const float* __restrict__ reg_b,
                             float* __restrict__ scores,
                             float* __restrict__ regs,
                             int npix) {
  __shared__ float dl[8][16][16];
  const int tilesPerBatch = (npix + 127) >> 7;
  const int b = blockIdx.x / tilesPerBatch;
  const int pixbase = (blockIdx.x % tilesPerBatch) << 7;
  const int lane = threadIdx.x & 31;
  const int wave = threadIdx.x >> 5;
  const int n0 = lane & 15;

  int pix = pixbase + (wave << 4) + n0; if (pix >= npix) pix = npix - 1;
  const unsigned* rowp = (const unsigned*)(x3 + ((size_t)b * npix + pix) * 32);
  int kb = (lane & 16) ? 8 : 0;
  union { v16h v; unsigned u[8]; } a;
  a.u[0] = rowp[(kb +  0) >> 1];
  a.u[1] = rowp[(kb +  2) >> 1];
  a.u[2] = rowp[(kb +  4) >> 1];
  a.u[3] = rowp[(kb +  6) >> 1];
  a.u[4] = rowp[(kb + 16) >> 1];
  a.u[5] = rowp[(kb + 18) >> 1];
  a.u[6] = rowp[(kb + 20) >> 1];
  a.u[7] = rowp[(kb + 22) >> 1];
  v16h bf = *(const v16h*)(Bh + ((size_t)lane << 4));

  float bv = (n0 < 2) ? cls_b[n0] : ((n0 < 6) ? reg_b[n0 - 2] : 0.f);
  union U8 { v8f v; float f[8]; } acc;
#pragma unroll
  for (int i = 0; i < 8; ++i) acc.f[i] = bv;
  acc.v = wmma16x16x32(a.v, bf, acc.v);

#pragma unroll
  for (int i = 0; i < 8; ++i) {
    int row = (lane & 16) ? 8 + i : i;
    dl[wave][row][n0] = acc.f[i];
  }
  __syncthreads();

  int p = threadIdx.x;
  if (p < 128) {
    int gpix = pixbase + p;
    if (gpix < npix) {
      const float* row = dl[p >> 4][p & 15];
      float c0 = row[0], c1 = row[1];
      float mx = fmaxf(c0, c1);
      float e0 = __expf(c0 - mx), e1 = __expf(c1 - mx);
      scores[(size_t)b * npix + gpix] = e1 / (e0 + e1);
      float* rp = regs + ((size_t)b * npix + gpix) * 4;
      rp[0] = row[2]; rp[1] = row[3]; rp[2] = row[4]; rp[3] = row[5];
    }
  }
}

// ---------------------------------------------------------------------------
// Per-batch top-1024 via incremental hierarchical argmax, all in LDS.
// locv/loci: per-thread max over its 62-elem stride; grpv/grpi: per-wave max.
// Per round: wave0 shfl-reduces the 32 group maxima; only the owner thread of
// the removed element rescans; only its wave refreshes the group max.
// Taken entries marked NaN (fail all compares). Stable lowest-index tie-break
// matches jax.lax.top_k. Winners buffered, emitted in parallel at the end.
// ---------------------------------------------------------------------------
__global__ void topk_select(const float* __restrict__ scores,
                            const float* __restrict__ regs,
                            float* __restrict__ vals,
                            float* __restrict__ boxes,
                            float* __restrict__ regsg,
                            int npix, int Wout) {
  extern __shared__ float tsm[];
  float* sc   = tsm;                        // [63104]
  float* locv = sc + 63104;                 // [1024]
  int*   loci = (int*)(locv + 1024);        // [1024]
  float* grpv = (float*)(loci + 1024);      // [32]
  int*   grpi = (int*)(grpv + 32);          // [32]
  float* wkv  = (float*)(grpi + 32);        // [1024]
  int*   wki  = (int*)(wkv + 1024);         // [1024]
  int*   bc   = wki + 1024;                 // [1]
  const int b = blockIdx.x, t = threadIdx.x;
  const int lane = t & 31, wave = t >> 5;
  const float NINF = -__builtin_inff();

  for (int i = t; i < npix; i += 1024) {
    float s = scores[(size_t)b * npix + i];
    sc[i] = (s > THR0_C) ? s : NINF;
  }
  __syncthreads();

  float bv = NINF; int bi = 0x7fffffff;
  for (int i = t; i < npix; i += 1024) {
    float v = sc[i];
    if (v > bv || (v == bv && i < bi)) { bv = v; bi = i; }
  }
  locv[t] = bv; loci[t] = bi;
  for (int d = 16; d > 0; d >>= 1) {
    float ov = __shfl_xor(bv, d); int oi = __shfl_xor(bi, d);
    if (ov > bv || (ov == bv && oi < bi)) { bv = ov; bi = oi; }
  }
  if (lane == 0) { grpv[wave] = bv; grpi[wave] = bi; }
  __syncthreads();

  for (int k = 0; k < K_TOP; ++k) {
    if (wave == 0) {
      float v = grpv[lane]; int i = grpi[lane];
      for (int d = 16; d > 0; d >>= 1) {
        float ov = __shfl_xor(v, d); int oi = __shfl_xor(i, d);
        if (ov > v || (ov == v && oi < i)) { v = ov; i = oi; }
      }
      if (lane == 0) {
        wkv[k] = v; wki[k] = i; bc[0] = i;
        sc[i] = __builtin_nanf("");          // consume winner
      }
    }
    __syncthreads();
    int widx = bc[0] & 1023;                 // owner thread of removed element
    if (t == widx) {
      float nv = NINF; int ni = 0x7fffffff;
      for (int i2 = t; i2 < npix; i2 += 1024) {
        float v2 = sc[i2];                   // NaN entries skipped
        if (v2 > nv || (v2 == nv && i2 < ni)) { nv = v2; ni = i2; }
      }
      locv[t] = nv; loci[t] = ni;
    }
    __syncthreads();
    int gw = widx >> 5;
    if (wave == gw) {
      float v = locv[(gw << 5) + lane]; int i = loci[(gw << 5) + lane];
      for (int d = 16; d > 0; d >>= 1) {
        float ov = __shfl_xor(v, d); int oi = __shfl_xor(i, d);
        if (ov > v || (ov == v && oi < i)) { v = ov; i = oi; }
      }
      if (lane == 0) { grpv[gw] = v; grpi[gw] = i; }
    }
    __syncthreads();
  }

  // Parallel emit: thread t materializes winner t.
  {
    float v = wkv[t]; int idx = wki[t];
    vals[b * K_TOP + t] = v;
    float jj = (float)(idx % Wout), ii = (float)(idx / Wout);
    float* bx = boxes + (size_t)(b * K_TOP + t) * 4;
    bx[0] = roundf(2.f * jj + 1.f);
    bx[1] = roundf(2.f * ii + 1.f);
    bx[2] = roundf(2.f * jj + 12.f);
    bx[3] = roundf(2.f * ii + 12.f);
    const float* rp = regs + ((size_t)b * npix + idx) * 4;
    float* rg = regsg + (size_t)(b * K_TOP + t) * 4;
    rg[0] = rp[0]; rg[1] = rp[1]; rg[2] = rp[2]; rg[3] = rp[3];
  }
}

// ---------------------------------------------------------------------------
// Bitmask NMS: phase 1 builds the 1024x1024 suppression bitmask in parallel
// (one row per thread, 32 u32 words in LDS); phase 2 is a single wave doing
// the sequential greedy cascade with shfl broadcast + bitwise OR (no barriers
// in the hot loop). Matches the reference fori_loop semantics exactly.
// ---------------------------------------------------------------------------
__global__ void nms_write(const float* __restrict__ vals,
                          const float* __restrict__ boxes,
                          const float* __restrict__ regsg,
                          float* __restrict__ out) {
  extern __shared__ unsigned char nsm[];
  float* X1 = (float*)nsm;             // [1024]
  float* Y1 = X1 + K_TOP;              // [1024]
  float* X2 = Y1 + K_TOP;              // [1024]
  float* Y2 = X2 + K_TOP;              // [1024]
  float* AR = Y2 + K_TOP;              // [1024]
  unsigned* SUPP = (unsigned*)(AR + K_TOP);   // [32]
  unsigned* MASK = SUPP + 32;                 // [1024*32]
  const int b = blockIdx.x, t = threadIdx.x;

  float v = vals[b * K_TOP + t];
  bool valid = (v > -1.0e30f);
  const float* bx = boxes + (size_t)(b * K_TOP + t) * 4;
  float x1 = bx[0], y1 = bx[1], x2 = bx[2], y2 = bx[3];
  float area = (x2 - x1 + 1.f) * (y2 - y1 + 1.f);
  X1[t] = x1; Y1[t] = y1; X2[t] = x2; Y2[t] = y2; AR[t] = area;
  if (t < 32) SUPP[t] = 0u;
  __syncthreads();
  if (!valid) atomicOr(&SUPP[t >> 5], 1u << (t & 31));  // invalid = pre-suppressed

  // Phase 1: row t of the suppression mask (t suppresses j if j>t & iou>thr)
  for (int w = 0; w < 32; ++w) {
    unsigned bits = 0u;
    int j0 = w << 5;
    if (j0 + 31 > t) {                 // skip all-zero words
#pragma unroll 4
      for (int bl = 0; bl < 32; ++bl) {
        int j = j0 + bl;
        if (j > t) {
          float xx1 = fmaxf(x1, X1[j]), yy1 = fmaxf(y1, Y1[j]);
          float xx2 = fminf(x2, X2[j]), yy2 = fminf(y2, Y2[j]);
          float ww = fmaxf(0.f, xx2 - xx1 + 1.f);
          float hh = fmaxf(0.f, yy2 - yy1 + 1.f);
          float inter = ww * hh;
          float iou = inter / (area + AR[j] - inter);
          if (iou > NMS_THR) bits |= (1u << bl);
        }
      }
    }
    MASK[t * 32 + w] = bits;
  }
  __syncthreads();

  // Phase 2: sequential greedy cascade, single wave, lane = column word.
  if (t < 32) {
    unsigned sup = SUPP[t];
    for (int i = 0; i < K_TOP; ++i) {
      unsigned wi = (unsigned)__shfl((int)sup, i >> 5);
      if (!((wi >> (i & 31)) & 1u)) sup |= MASK[i * 32 + t];
    }
    SUPP[t] = sup;
  }
  __syncthreads();

  bool keep = !((SUPP[t >> 5] >> (t & 31)) & 1u);   // false for invalid too
  bool on = keep && valid;
  const float* rg = regsg + (size_t)(b * K_TOP + t) * 4;
  float* op = out + (size_t)(b * K_TOP + t) * 9;
  op[0] = on ? x1 : 0.f;
  op[1] = on ? y1 : 0.f;
  op[2] = on ? x2 : 0.f;
  op[3] = on ? y2 : 0.f;
  op[4] = on ? v  : 0.f;
  op[5] = on ? rg[0] : 0.f;
  op[6] = on ? rg[1] : 0.f;
  op[7] = on ? rg[2] : 0.f;
  op[8] = on ? rg[3] : 0.f;
}

// ---------------------------------------------------------------------------
extern "C" void kernel_launch(void* const* d_in, const int* in_sizes, int n_in,
                              void* d_out, int out_size, void* d_ws, size_t ws_size,
                              hipStream_t stream) {
  (void)in_sizes; (void)n_in; (void)out_size; (void)ws_size;
  const float* img  = (const float*)d_in[0];
  const float* w1   = (const float*)d_in[1];
  const float* b1   = (const float*)d_in[2];
  const float* a1   = (const float*)d_in[3];
  const float* w2   = (const float*)d_in[4];
  const float* b2   = (const float*)d_in[5];
  const float* a2   = (const float*)d_in[6];
  const float* w3   = (const float*)d_in[7];
  const float* b3   = (const float*)d_in[8];
  const float* a3   = (const float*)d_in[9];
  const float* clsw = (const float*)d_in[10];
  const float* clsb = (const float*)d_in[11];
  const float* regw = (const float*)d_in[12];
  const float* regb = (const float*)d_in[13];
  float* out = (float*)d_out;

  const int B = 16;
  const int NP1 = 255 * 255;   // after conv1+pool
  const int NP2 = 253 * 253;   // after conv2
  const int NP3 = 251 * 251;   // after conv3

  char* ws = (char*)d_ws;
  size_t off = 0;
  auto carve = [&](size_t bytes) -> void* {
    void* p = ws + off;
    off += (bytes + 255) & ~(size_t)255;
    return p;
  };
  _Float16* p1    = (_Float16*)carve((size_t)B * NP1 * 10 * 2);
  _Float16* x2    = (_Float16*)carve((size_t)B * NP2 * 16 * 2);
  _Float16* x3    = (_Float16*)carve((size_t)B * NP3 * 32 * 2);
  _Float16* Bp2   = (_Float16*)carve((size_t)1 * 3 * 512 * 2);
  _Float16* Bp3   = (_Float16*)carve((size_t)2 * 5 * 512 * 2);
  _Float16* Bph   = (_Float16*)carve((size_t)512 * 2);
  float* scores   = (float*)carve((size_t)B * NP3 * 4);
  float* regs     = (float*)carve((size_t)B * NP3 * 16);
  float* vals     = (float*)carve((size_t)B * K_TOP * 4);
  float* boxes    = (float*)carve((size_t)B * K_TOP * 16);
  float* regsg    = (float*)carve((size_t)B * K_TOP * 16);

  // Weight packing (tiny)
  pack_conv_b<<<(1 * 3 * 512 + 255) / 256, 256, 0, stream>>>(w2, Bp2, 10, 16, 1, 3);
  pack_conv_b<<<(2 * 5 * 512 + 255) / 256, 256, 0, stream>>>(w3, Bp3, 16, 32, 2, 5);
  pack_head_b<<<2, 256, 0, stream>>>(clsw, regw, Bph);

  // Stage 1: conv1 + prelu + maxpool (streams the 50MB input once)
  conv1_prelu_pool<<<(B * NP1 + 255) / 256, 256, 0, stream>>>(img, w1, b1, a1, p1);

  // Stage 2: conv2 (K=90 -> 3 chunks of 32, Kpad=96), 16 out ch
  {
    int tiles = (NP2 + 127) / 128;
    conv3x3_wmma<1><<<B * tiles, 256, 128 * 96 * 2, stream>>>(
        p1, Bp2, b2, a2, x2, 255, 255, 10, 253, 253, 16, 3, 96);
  }
  // Stage 3: conv3 (K=144 -> 5 chunks, Kpad=160), 32 out ch (2 N-tiles)
  {
    int tiles = (NP3 + 127) / 128;
    conv3x3_wmma<2><<<B * tiles, 256, 128 * 160 * 2, stream>>>(
        x2, Bp3, b3, a3, x3, 253, 253, 16, 251, 251, 32, 5, 160);
  }
  // Stage 4: 1x1 heads + softmax score
  {
    int tiles = (NP3 + 127) / 128;
    head_softmax<<<B * tiles, 256, 0, stream>>>(x3, Bph, clsb, regb, scores, regs, NP3);
  }
  // Stage 5: per-batch top-1024 (scores + caches resident in 320KB LDS)
  {
    size_t sm = (size_t)(63104 + 1024 + 1024 + 32 + 32 + 1024 + 1024 + 1) * 4;
    topk_select<<<B, 1024, sm, stream>>>(scores, regs, vals, boxes, regsg, NP3, 251);
  }
  // Stage 6: bitmask NMS + masked output
  {
    size_t sm = (size_t)(5 * K_TOP) * 4 + (size_t)(32 + K_TOP * 32) * 4;
    nms_write<<<B, 1024, sm, stream>>>(vals, boxes, regsg, out);
  }
}
